// P6TopkCapGate_773094113557
// MI455X (gfx1250) — matrix operand order repaired and checked by reading, not compiled, not run on hardware
//
#include <hip/hip_runtime.h>

// ---------------------------------------------------------------------------
// MoE router gate for MI455X (gfx1250, wave32, WMMA).
//   logits = hidden[8192,4096] @ (0.5*(wg+wg_ema))[4096,64]
// GEMM via split-bf16: f32 = hi(bf16) + lo(bf16 residual);
//   A@B ~= Ahi@Bhi + Ahi@Blo + Alo@Bhi   (3x v_wmma_f32_16x16x32_bf16 / K=32)
// Three independent accumulator chains -> no WMMA->WMMA RAW stalls in the loop.
// ---------------------------------------------------------------------------

typedef __attribute__((ext_vector_type(16))) __bf16 v16bf;
typedef __attribute__((ext_vector_type(8)))  float  v8f;
typedef __attribute__((ext_vector_type(4)))  float  v4f;

constexpr int T_DIM = 8192;
constexpr int H_DIM = 4096;
constexpr int E_DIM = 64;

constexpr int BM   = 32;   // rows per block
constexpr int KC   = 64;   // K chunk staged in LDS
constexpr int ASTR = 70;   // LDS row stride (bf16 elems); 35 dwords -> conflict-free

// output regions (flat f32 elements)
constexpr size_t OFF_TOPK   = 0;
constexpr size_t OFF_LOGITS = OFF_TOPK + (size_t)T_DIM * 2;          // 16384
constexpr size_t OFF_MASK   = OFF_LOGITS + (size_t)T_DIM * E_DIM;    // 540672
constexpr size_t OFF_SEL    = OFF_MASK + (size_t)E_DIM * 2 * T_DIM;  // 1589248

__device__ __forceinline__ unsigned short bf_bits(__bf16 h) {
  return __builtin_bit_cast(unsigned short, h);
}

// ---- Kernel 1: wg_running = 0.5*(wg+wg_ema), split to bf16 hi/lo, transposed [E][H]
__global__ __launch_bounds__(256) void prep_kernel(const float* __restrict__ wg,
                                                   const float* __restrict__ wge,
                                                   __bf16* __restrict__ bhi,
                                                   __bf16* __restrict__ blo) {
  int i = (blockIdx.x * 256 + threadIdx.x) * 4;      // 256 blocks -> 262144 elems
  v4f a = *reinterpret_cast<const v4f*>(wg + i);
  v4f b = *reinterpret_cast<const v4f*>(wge + i);
  int k = i >> 6;        // row in [H]
  int e = i & 63;        // expert base (aligned to 4)
#pragma unroll
  for (int j = 0; j < 4; ++j) {
    float w   = 0.5f * (a[j] + b[j]);
    __bf16 h  = (__bf16)w;
    __bf16 l  = (__bf16)(w - (float)h);
    bhi[(size_t)(e + j) * H_DIM + k] = h;
    blo[(size_t)(e + j) * H_DIM + k] = l;
  }
}

// ---- Kernel 2: zero the expert-mask region
__global__ __launch_bounds__(256) void zero_kernel(float* __restrict__ p) {
  int i = (blockIdx.x * 256 + threadIdx.x) * 4;
  *reinterpret_cast<v4f*>(p + i) = (v4f){0.f, 0.f, 0.f, 0.f};
}

// ---- Kernel 3: split-bf16 WMMA GEMM -> router_logits [T,64]
__global__ __launch_bounds__(256) void gemm_kernel(const float*  __restrict__ A,
                                                   const __bf16* __restrict__ Bhi,
                                                   const __bf16* __restrict__ Blo,
                                                   float* __restrict__ L) {
  __shared__ __bf16 sAhi[BM * ASTR];
  __shared__ __bf16 sAlo[BM * ASTR];

  const int tid  = threadIdx.x;
  const int wave = tid >> 5;
  const int lane = tid & 31;
  const int g    = lane >> 4;    // lane half
  const int mn   = lane & 15;    // M (for A/C) or N (for B/C)

  const int rowBase = blockIdx.x * BM;
  const int rowT    = (wave >> 2) * 16;   // 0 or 16
  const int colT    = (wave & 3) * 16;    // 0,16,32,48

  // three independent accumulation chains (hi*hi, hi*lo, lo*hi)
  v8f acc0 = {0.f, 0.f, 0.f, 0.f, 0.f, 0.f, 0.f, 0.f};
  v8f acc1 = {0.f, 0.f, 0.f, 0.f, 0.f, 0.f, 0.f, 0.f};
  v8f acc2 = {0.f, 0.f, 0.f, 0.f, 0.f, 0.f, 0.f, 0.f};

  // A loader mapping: 2048 floats per chunk / 256 threads = 8 floats each
  const int arow = tid >> 3;
  const int aoff = (tid & 7) * 8;
  const float* aptr = A + (size_t)(rowBase + arow) * H_DIM + aoff;

  // B fragment base: transposed layout makes a fragment 16 contiguous bf16 (32B)
  const __bf16* bh = Bhi + (size_t)(colT + mn) * H_DIM + g * 16;
  const __bf16* bl = Blo + (size_t)(colT + mn) * H_DIM + g * 16;

  unsigned int* sAhiU = reinterpret_cast<unsigned int*>(sAhi);
  unsigned int* sAloU = reinterpret_cast<unsigned int*>(sAlo);

  for (int k0 = 0; k0 < H_DIM; k0 += KC) {
    __syncthreads();                       // previous chunk fully consumed
    if (k0 + KC < H_DIM) __builtin_prefetch(aptr + k0 + KC, 0, 2);
    v4f a0 = *reinterpret_cast<const v4f*>(aptr + k0);
    v4f a1 = *reinterpret_cast<const v4f*>(aptr + k0 + 4);
    float av[8] = {a0[0], a0[1], a0[2], a0[3], a1[0], a1[1], a1[2], a1[3]};
    const int ubase = (arow * ASTR + aoff) >> 1;   // dword index (aoff even)
#pragma unroll
    for (int i = 0; i < 4; ++i) {
      float  w0 = av[2 * i], w1 = av[2 * i + 1];
      __bf16 h0 = (__bf16)w0, h1 = (__bf16)w1;
      __bf16 l0 = (__bf16)(w0 - (float)h0), l1 = (__bf16)(w1 - (float)h1);
      sAhiU[ubase + i] = ((unsigned int)bf_bits(h1) << 16) | bf_bits(h0);
      sAloU[ubase + i] = ((unsigned int)bf_bits(l1) << 16) | bf_bits(l0);
    }
    __syncthreads();

#pragma unroll
    for (int kk = 0; kk < KC; kk += 32) {
      // A fragments (contiguous K layout: vgpr v holds K = g*16+v*2, +1)
      v16bf ahi, alo;
      unsigned int* ah = reinterpret_cast<unsigned int*>(&ahi);
      unsigned int* al = reinterpret_cast<unsigned int*>(&alo);
      const unsigned int* sh =
          reinterpret_cast<const unsigned int*>(sAhi + (rowT + mn) * ASTR + kk);
      const unsigned int* sl =
          reinterpret_cast<const unsigned int*>(sAlo + (rowT + mn) * ASTR + kk);
#pragma unroll
      for (int v = 0; v < 8; ++v) {
        int K = (g << 4) + (v << 1);  // even
        ah[v] = sh[K >> 1];
        al[v] = sl[K >> 1];
      }
      // B fragments: lane half g covers K = kk + g*16 + [0..15], contiguous in ws
      v16bf fbh = *reinterpret_cast<const v16bf*>(bh + k0 + kk);
      v16bf fbl = *reinterpret_cast<const v16bf*>(bl + k0 + kk);

      acc0 = __builtin_amdgcn_wmma_f32_16x16x32_bf16(false, ahi, false, fbh,
                                                     (short)0, acc0, false, false);
      acc1 = __builtin_amdgcn_wmma_f32_16x16x32_bf16(false, ahi, false, fbl,
                                                     (short)0, acc1, false, false);
      acc2 = __builtin_amdgcn_wmma_f32_16x16x32_bf16(false, alo, false, fbh,
                                                     (short)0, acc2, false, false);
    }
  }

  // combine the three chains; C/D layout: vgpr r -> M = r + 8*g, N = mn
  float* Lp = L + (size_t)(rowBase + rowT + g * 8) * E_DIM + colT + mn;
#pragma unroll
  for (int r = 0; r < 8; ++r)
    Lp[(size_t)r * E_DIM] = acc0[r] + (acc1[r] + acc2[r]);
}

// ---- Kernel 4: per-row top-2 over logits + renormalized softmax weights + scatter
__global__ __launch_bounds__(256) void topk_kernel(const float* __restrict__ logits,
                                                   float* __restrict__ topkw,
                                                   float* __restrict__ mask,
                                                   float* __restrict__ sel) {
  const int row  = blockIdx.x * 8 + (threadIdx.x >> 5);   // one wave32 per row
  const int lane = threadIdx.x & 31;
  const float* lr = logits + (size_t)row * E_DIM;
  float l0 = lr[lane], l1 = lr[lane + 32];

  float v1, v2; int i1, i2;
  if (l0 >= l1) { v1 = l0; i1 = lane;      v2 = l1; i2 = lane + 32; }
  else          { v1 = l1; i1 = lane + 32; v2 = l0; i2 = lane; }

#pragma unroll
  for (int m = 16; m >= 1; m >>= 1) {
    float pv1 = __shfl_xor(v1, m, 32);
    int   pi1 = __shfl_xor(i1, m, 32);
    float pv2 = __shfl_xor(v2, m, 32);
    int   pi2 = __shfl_xor(i2, m, 32);
    bool aW = (v1 > pv1) || (v1 == pv1 && i1 < pi1);
    float n1v = aW ? v1 : pv1;  int n1i = aW ? i1 : pi1;
    float c2v = aW ? pv1 : v1;  int c2i = aW ? pi1 : i1;  // loser of top compare
    float d2v = aW ? v2 : pv2;  int d2i = aW ? i2 : pi2;  // winner's runner-up
    bool dW = (d2v > c2v) || (d2v == c2v && d2i < c2i);
    v2 = dW ? d2v : c2v;  i2 = dW ? d2i : c2i;
    v1 = n1v;             i1 = n1i;
  }

  // 2-way softmax on top-2 logits == softmax-then-renormalize of the reference
  float t  = __expf(v2 - v1);         // <= 1
  float w1 = 1.0f / (1.0f + t);
  float w2 = t * w1;

  if (lane == 0) {
    topkw[(size_t)row * 2 + 0] = w1;
    topkw[(size_t)row * 2 + 1] = w2;
    sel[(size_t)row * 2 + 0] = (float)i1;
    sel[(size_t)row * 2 + 1] = (float)i2;
    // expert_mask[e][j][t]
    mask[(size_t)i1 * (2 * T_DIM) + 0 * T_DIM + row] = 1.0f;
    mask[(size_t)i2 * (2 * T_DIM) + 1 * T_DIM + row] = 1.0f;
  }
}

extern "C" void kernel_launch(void* const* d_in, const int* in_sizes, int n_in,
                              void* d_out, int out_size, void* d_ws, size_t ws_size,
                              hipStream_t stream) {
  const float* hidden = (const float*)d_in[0];
  const float* wg     = (const float*)d_in[1];
  const float* wg_ema = (const float*)d_in[2];
  float* out = (float*)d_out;

  // workspace: split-bf16 transposed gate weights (2 * 512 KB)
  __bf16* bhi = (__bf16*)d_ws;
  __bf16* blo = bhi + (size_t)H_DIM * E_DIM;

  prep_kernel<<<256, 256, 0, stream>>>(wg, wg_ema, bhi, blo);
  zero_kernel<<<1024, 256, 0, stream>>>(out + OFF_MASK);
  gemm_kernel<<<T_DIM / BM, 256, 0, stream>>>(hidden, bhi, blo, out + OFF_LOGITS);
  topk_kernel<<<T_DIM / 8, 256, 0, stream>>>(out + OFF_LOGITS, out + OFF_TOPK,
                                             out + OFF_MASK, out + OFF_SEL);
}
